// DecoderRNN_88845693485805
// MI455X (gfx1250) — matrix operand order repaired and checked by reading, not compiled, not run on hardware
//
#include <hip/hip_runtime.h>
#include <hip/hip_bf16.h>
#include <math.h>
#include <stdint.h>

typedef __attribute__((ext_vector_type(16))) _Float16 v16h;
typedef __attribute__((ext_vector_type(8)))  _Float16 v8h;
typedef __attribute__((ext_vector_type(8)))  float    v8f;
typedef __attribute__((ext_vector_type(4)))  float    v4f;
typedef __attribute__((ext_vector_type(4)))  unsigned int v4u;
typedef __attribute__((ext_vector_type(8)))  int      v8i;
typedef __attribute__((ext_vector_type(4)))  int      v4i;

#define BB   64
#define NF   196
#define ENC  2048
#define DEC  512
#define ATT  512
#define VOC  30000
#define EMB  512
#define TT   24

#if defined(__has_builtin)
#if __has_builtin(__builtin_amdgcn_tensor_load_to_lds) && __has_builtin(__builtin_amdgcn_s_wait_tensorcnt)
#define HAVE_TDM 1
#endif
#endif
#ifndef HAVE_TDM
#define HAVE_TDM 0
#endif

// ---------------------------------------------------------------------------
// One-time weight conversion f32 -> f16 WITH transpose: src (K,N) row-major
// becomes dst (N,K) "K-major" -> every GEMM B-tile is a contiguous-K 2-D
// block: ideal for TDM descriptors and vectorized LDS reads.
// ---------------------------------------------------------------------------
__global__ void cvt_transpose_f16(const float* __restrict__ src,
                                  _Float16* __restrict__ dst, int K, int N) {
    int i = blockIdx.x * blockDim.x + threadIdx.x;
    if (i < K * N) {
        int n = i / K, k = i % K;
        dst[i] = (_Float16)src[(size_t)k * N + n];
    }
}

// ---------------------------------------------------------------------------
// mean over feature axis
// ---------------------------------------------------------------------------
__global__ void mean_feat_kernel(const float* __restrict__ features,
                                 float* __restrict__ meanf) {
    int b = blockIdx.x;
    for (int e = threadIdx.x; e < ENC; e += blockDim.x) {
        float s = 0.f;
        const float* p = features + (size_t)b * NF * ENC + e;
        for (int f = 0; f < NF; ++f) s += p[(size_t)f * ENC];
        meanf[b * ENC + e] = s * (1.0f / NF);
    }
}

#if HAVE_TDM
// ---------------------------------------------------------------------------
// Tensor Data Mover: async-load a (64 x 128) f16 tile, X = 64 contiguous K
// elements, Y = 128 N-rows with stride K.  D# per CDNA5 ISA §8.3/§8.4.
// rem_k/rem_n give tensor extents from the tile origin -> OOB reads return 0.
// ---------------------------------------------------------------------------
__device__ __forceinline__ void tdm_load_b_tile(const _Float16* gsrc,
                                                unsigned lds_off,
                                                unsigned rem_k, unsigned rem_n,
                                                unsigned strideK) {
    unsigned long long ga = (unsigned long long)(uintptr_t)gsrc;
    v4u g0;
    g0[0] = 1u;                                            // count=1, user mode
    g0[1] = lds_off;                                       // lds_addr (bytes)
    g0[2] = (unsigned)ga;                                  // global_addr lo
    g0[3] = (unsigned)((ga >> 32) & 0x01FFFFFFull)         // global_addr[56:32]
          | (2u << 30);                                    // type = 2 ("image")
    v8i g1;
    g1[0] = (int)(1u << 16);                               // data_size = 2 bytes
    g1[1] = (int)((rem_k & 0xFFFFu) << 16);                // tensor_dim0 lo16
    g1[2] = (int)((rem_k >> 16) | ((rem_n & 0xFFFFu) << 16)); // dim0 hi / dim1 lo
    g1[3] = (int)((rem_n >> 16) | (64u << 16));            // dim1 hi / tile_dim0=64
    g1[4] = (int)128u;                                     // tile_dim1=128, tile_dim2=0
    g1[5] = (int)strideK;                                  // tensor_dim0_stride lo32
    g1[6] = 0;                                             // stride hi / dim1_stride
    g1[7] = 0;
    v4i gz4 = {0, 0, 0, 0};                                // groups 2/3 unused (2-D)
    v8i gz8 = {0, 0, 0, 0, 0, 0, 0, 0};
    __builtin_amdgcn_tensor_load_to_lds(g0, g1, gz4, gz4, gz8, 0);
}
#endif

// build a v16h fragment from two aligned 16-byte LDS chunks
__device__ __forceinline__ v16h frag16(const _Float16* p0, const _Float16* p1) {
    v8h lo = *(const v8h*)p0;
    v8h hi = *(const v8h*)p1;
    return __builtin_shufflevector(lo, hi, 0, 1, 2, 3, 4, 5, 6, 7,
                                   8, 9, 10, 11, 12, 13, 14, 15);
}

// ---------------------------------------------------------------------------
// WMMA GEMM:  C[M,N] = A[M,K](f32) * Bt(N,K)(f16, K-major) + bias (+C if accum)
// 4 waves/block; block tile 16(M) x 128(N); wave owns 2 n-tiles; K-step 64
// => 4 x v_wmma_f32_16x16x32_f16 per iteration.  B tiles staged to LDS
// (col-major, contiguous K per column) by the TDM, double-buffered and
// overlapped with compute; A tiles converted f32->f16 while staged.
// All fragment LDS traffic is explicit 16-byte vectors (ds_load_b128).
// Requires M%16==0, K%64==0.
// ---------------------------------------------------------------------------
__global__ void gemm_wmma_f16(const float* __restrict__ A, int lda,
                              const _Float16* __restrict__ Bt,   // (N,K)
                              const float* __restrict__ bias,
                              float* __restrict__ C, int ldc,
                              int M, int N, int K, int accumulate) {
    __shared__ __attribute__((aligned(32))) _Float16 As[16 * 64];     // 2 KB
    __shared__ __attribute__((aligned(32))) _Float16 Bs[2][128 * 64]; // 2x16 KB

    const int tid  = threadIdx.x;             // 128 threads = 4 waves
    const int wave = tid >> 5;
    const int lane = tid & 31;
    const int mt   = blockIdx.y;
    const int nc0  = blockIdx.x * 128;
    const int wc0  = nc0 + wave * 32;         // wave covers 32 columns

    const int nn = lane & 15;
    const int hi = lane >> 4;

    v8f acc0 = {0.f, 0.f, 0.f, 0.f, 0.f, 0.f, 0.f, 0.f};
    v8f acc1 = {0.f, 0.f, 0.f, 0.f, 0.f, 0.f, 0.f, 0.f};

    const int nIter = K >> 6;

#if HAVE_TDM
    if (wave == 0) {
        tdm_load_b_tile(Bt + (size_t)nc0 * K,
                        (unsigned)(uintptr_t)&Bs[0][0],
                        (unsigned)K, (unsigned)(N - nc0), (unsigned)K);
    }
#endif

    for (int it = 0; it < nIter; ++it) {
        const int k0 = it << 6;
        const int cur = it & 1;

        // stage A tile 16x64 f32->f16: thread owns 8 contiguous K elements
        // -> 2x global b128 loads + 1x ds_store_b128
        {
            int idx = tid * 8;                 // 128 threads * 8 = 1024 elems
            int r = idx >> 6, kk = idx & 63;
            const float* src = &A[(size_t)(mt * 16 + r) * lda + (k0 + kk)];
            v4f q0 = *(const v4f*)(src);
            v4f q1 = *(const v4f*)(src + 4);
            v8h h8;
#pragma unroll
            for (int j = 0; j < 4; ++j) {
                h8[j]     = (_Float16)q0[j];
                h8[4 + j] = (_Float16)q1[j];
            }
            *(v8h*)&As[r * 64 + kk] = h8;
        }

#if HAVE_TDM
        if (wave == 0) {
            if (it + 1 < nIter) {
                tdm_load_b_tile(Bt + (size_t)nc0 * K + (k0 + 64),
                                (unsigned)(uintptr_t)&Bs[cur ^ 1][0],
                                (unsigned)(K - k0 - 64), (unsigned)(N - nc0),
                                (unsigned)K);
                __builtin_amdgcn_s_wait_tensorcnt(1);   // current tile landed
            } else {
                __builtin_amdgcn_s_wait_tensorcnt(0);
            }
        }
#else
        // manual staging fallback: thread t copies column (nc0+t)'s 64-K run
        {
            int col = nc0 + tid;
            v8h* dstc = (v8h*)&Bs[cur][tid * 64];
            if (col < N) {
                const _Float16* srcc = Bt + (size_t)col * K + k0;
#pragma unroll
                for (int v = 0; v < 8; ++v) dstc[v] = *(const v8h*)(srcc + v * 8);
            } else {
                v8h z = {};
#pragma unroll
                for (int v = 0; v < 8; ++v) dstc[v] = z;
            }
        }
#endif
        __syncthreads();

#pragma unroll
        for (int sk = 0; sk < 2; ++sk) {
            // A frag: rows = lane&15; K = sk*32 + 16*(e/8) + 8*hi + (e%8)
            const _Float16* ap = &As[nn * 64 + (sk << 5) + (hi << 3)];
            v16h af = frag16(ap, ap + 16);
            // B frag: col-major LDS -> K = sk*32 + 16*(e/8) + 8*hi + (e%8)
            const _Float16* bp0 = &Bs[cur][(wave * 32 + nn) * 64 + (sk << 5) + (hi << 3)];
            const _Float16* bp1 = bp0 + 16 * 64;
            v16h bf0 = frag16(bp0, bp0 + 16);
            v16h bf1 = frag16(bp1, bp1 + 16);
            acc0 = __builtin_amdgcn_wmma_f32_16x16x32_f16(
                false, af, false, bf0, (short)0, acc0, false, false);
            acc1 = __builtin_amdgcn_wmma_f32_16x16x32_f16(
                false, af, false, bf1, (short)0, acc1, false, false);
        }
        __syncthreads();
    }

    // epilogue: bias add (+ optional accumulate), per-sub-tile N guard
#pragma unroll
    for (int s = 0; s < 2; ++s) {
        int tcol = wc0 + s * 16;
        if (tcol < N) {
            int n = tcol + nn;
            float bv = bias ? bias[n] : 0.f;
            const v8f& a = s ? acc1 : acc0;
#pragma unroll
            for (int r = 0; r < 8; ++r) {
                int row = mt * 16 + hi * 8 + r;       // D layout: M = 8*hi + r
                float* p = &C[(size_t)row * ldc + n];
                float v = a[r] + bv;
                if (accumulate) v += *p;
                *p = v;
            }
        }
    }
}

// ---------------------------------------------------------------------------
// Attention + embedding gather for step t (one block per batch element).
// ---------------------------------------------------------------------------
__global__ void attention_step_kernel(const float* __restrict__ features,
                                      const float* __restrict__ uhs,
                                      const float* __restrict__ h,
                                      const float* __restrict__ Ww,
                                      const float* __restrict__ Wb,
                                      const float* __restrict__ Aw,
                                      const float* __restrict__ Ab,
                                      const float* __restrict__ emb,
                                      const int*   __restrict__ captions,
                                      float* __restrict__ Astep,
                                      float* __restrict__ alphas_out,
                                      int t) {
    __shared__ float wah[ATT];
    __shared__ float sc[NF];
    __shared__ float red[2];

    int b = blockIdx.x;
    int tid = threadIdx.x;

    for (int a = tid; a < ATT; a += blockDim.x) {
        float s = Wb[a];
        const float* hb = h + b * DEC;
        for (int k = 0; k < DEC; ++k) s += hb[k] * Ww[k * ATT + a];
        wah[a] = s;
    }
    int tok = captions[b * (TT + 1) + t];
    for (int e = tid; e < EMB; e += blockDim.x)
        Astep[(size_t)b * (EMB + ENC) + e] = emb[(size_t)tok * EMB + e];
    __syncthreads();

    if (tid < NF) {
        const float* u = uhs + ((size_t)b * NF + tid) * ATT;
        float s = 0.f;
        for (int a = 0; a < ATT; ++a) s += tanhf(u[a] + wah[a]) * Aw[a];
        sc[tid] = s + Ab[0];
    }
    __syncthreads();

    if (tid == 0) {
        float m = sc[0];
        for (int f = 1; f < NF; ++f) m = fmaxf(m, sc[f]);
        float sum = 0.f;
        for (int f = 0; f < NF; ++f) sum += expf(sc[f] - m);
        red[0] = m; red[1] = sum;
    }
    __syncthreads();
    if (tid < NF) {
        float al = expf(sc[tid] - red[0]) / red[1];
        sc[tid] = al;
        alphas_out[((size_t)b * TT + t) * NF + tid] = al;
    }
    __syncthreads();

    for (int e = tid; e < ENC; e += blockDim.x) {
        float s = 0.f;
        const float* p = features + (size_t)b * NF * ENC + e;
        for (int f = 0; f < NF; ++f) s += sc[f] * p[(size_t)f * ENC];
        Astep[(size_t)b * (EMB + ENC) + EMB + e] = s;
    }
}

// ---------------------------------------------------------------------------
// LSTM pointwise update
// ---------------------------------------------------------------------------
__device__ __forceinline__ float sigm(float x) { return 1.f / (1.f + expf(-x)); }

__global__ void lstm_update_kernel(const float* __restrict__ gates,
                                   float* __restrict__ h,
                                   float* __restrict__ c) {
    int idx = blockIdx.x * blockDim.x + threadIdx.x;
    if (idx >= BB * DEC) return;
    int b = idx / DEC, j = idx % DEC;
    const float* g = gates + (size_t)b * 4 * DEC;
    float gi = g[j], gf = g[DEC + j], gg = g[2 * DEC + j], go = g[3 * DEC + j];
    float cn = sigm(gf) * c[idx] + sigm(gi) * tanhf(gg);
    c[idx] = cn;
    h[idx] = sigm(go) * tanhf(cn);
}

// ---------------------------------------------------------------------------
// Host orchestration
// ---------------------------------------------------------------------------
extern "C" void kernel_launch(void* const* d_in, const int* in_sizes, int n_in,
                              void* d_out, int out_size, void* d_ws, size_t ws_size,
                              hipStream_t stream) {
    const float* features = (const float*)d_in[0];
    const int*   captions = (const int*)  d_in[1];
    const float* emb      = (const float*)d_in[2];
    const float* Uw       = (const float*)d_in[3];
    const float* Ub       = (const float*)d_in[4];
    const float* Ww       = (const float*)d_in[5];
    const float* Wb       = (const float*)d_in[6];
    const float* Aw       = (const float*)d_in[7];
    const float* Ab       = (const float*)d_in[8];
    const float* ih_w     = (const float*)d_in[9];
    const float* ih_b     = (const float*)d_in[10];
    const float* ic_w     = (const float*)d_in[11];
    const float* ic_b     = (const float*)d_in[12];
    const float* W_ih     = (const float*)d_in[13];
    const float* b_ih     = (const float*)d_in[14];
    const float* W_hh     = (const float*)d_in[15];
    const float* b_hh     = (const float*)d_in[16];
    const float* fcn_w    = (const float*)d_in[17];
    const float* fcn_b    = (const float*)d_in[18];

    float* preds  = (float*)d_out;                            // (64,24,30000)
    float* alphas = (float*)d_out + (size_t)BB * TT * VOC;    // (64,24,196)

    char* ws = (char*)d_ws;
    size_t off = 0;
    auto carve = [&](size_t bytes) -> void* {
        void* p = ws + off;
        off += (bytes + 255) & ~(size_t)255;
        return p;
    };
    float*    uhs    = (float*)   carve((size_t)BB * NF * ATT * 4);
    _Float16* Uw16   = (_Float16*)carve((size_t)ENC * ATT * 2);            // (ATT,ENC)
    _Float16* ihw16  = (_Float16*)carve((size_t)ENC * DEC * 2);            // (DEC,ENC)
    _Float16* icw16  = (_Float16*)carve((size_t)ENC * DEC * 2);
    _Float16* Wih16  = (_Float16*)carve((size_t)(EMB + ENC) * 4 * DEC * 2);// (2048,2560)
    _Float16* Whh16  = (_Float16*)carve((size_t)DEC * 4 * DEC * 2);        // (2048,512)
    _Float16* fcn16  = (_Float16*)carve((size_t)DEC * VOC * 2);            // (30000,512)
    float*    meanf  = (float*)   carve((size_t)BB * ENC * 4);
    float*    hbuf   = (float*)   carve((size_t)BB * DEC * 4);
    float*    cbuf   = (float*)   carve((size_t)BB * DEC * 4);
    float*    Astep  = (float*)   carve((size_t)BB * (EMB + ENC) * 4);
    float*    gates  = (float*)   carve((size_t)BB * 4 * DEC * 4);
    (void)ws_size; (void)n_in; (void)in_sizes; (void)out_size;

    auto cvtT = [&](const float* s, _Float16* d, int K, int N) {
        int n = K * N;
        cvt_transpose_f16<<<(n + 255) / 256, 256, 0, stream>>>(s, d, K, N);
    };
    auto gemm = [&](const float* A, int lda, const _Float16* Bm,
                    const float* bias, float* C, int ldc,
                    int M, int N, int K, int accum) {
        dim3 grid((N + 127) / 128, M / 16);
        gemm_wmma_f16<<<grid, 128, 0, stream>>>(A, lda, Bm, bias, C, ldc,
                                                M, N, K, accum);
    };

    // one-time weight conversions: f32 (K,N) -> f16 (N,K)
    cvtT(Uw,    Uw16,  ENC, ATT);
    cvtT(ih_w,  ihw16, ENC, DEC);
    cvtT(ic_w,  icw16, ENC, DEC);
    cvtT(W_ih,  Wih16, EMB + ENC, 4 * DEC);
    cvtT(W_hh,  Whh16, DEC, 4 * DEC);
    cvtT(fcn_w, fcn16, DEC, VOC);

    mean_feat_kernel<<<BB, 256, 0, stream>>>(features, meanf);

    // u_hs = features @ Uw + Ub     (12544 x 512, K=2048)
    gemm(features, ENC, Uw16, Ub, uhs, ATT, BB * NF, ATT, ENC, 0);
    // h = mean_f @ ih_w + ih_b ; c = mean_f @ ic_w + ic_b
    gemm(meanf, ENC, ihw16, ih_b, hbuf, DEC, BB, DEC, ENC, 0);
    gemm(meanf, ENC, icw16, ic_b, cbuf, DEC, BB, DEC, ENC, 0);

    for (int t = 0; t < TT; ++t) {
        attention_step_kernel<<<BB, 256, 0, stream>>>(
            features, uhs, hbuf, Ww, Wb, Aw, Ab, emb, captions,
            Astep, alphas, t);
        // gates = [x_t, ctx] @ W_ih + b_ih     (64 x 2048, K=2560)
        gemm(Astep, EMB + ENC, Wih16, b_ih, gates, 4 * DEC,
             BB, 4 * DEC, EMB + ENC, 0);
        // gates += h @ W_hh + b_hh             (64 x 2048, K=512)
        gemm(hbuf, DEC, Whh16, b_hh, gates, 4 * DEC,
             BB, 4 * DEC, DEC, 1);
        lstm_update_kernel<<<(BB * DEC + 255) / 256, 256, 0, stream>>>(
            gates, hbuf, cbuf);
        // preds[:, t, :] = h @ fcn_w + fcn_b   (64 x 30000, K=512)
        gemm(hbuf, DEC, fcn16, fcn_b, preds + (size_t)t * VOC,
             TT * VOC, BB, VOC, DEC, 0);
    }
}
